// MambaLayer_3745211482475
// MI455X (gfx1250) — compile-verified, matrix operand order checked by reading, not run on hardware
//
#include <hip/hip_runtime.h>
#include <hip/hip_bf16.h>

// ---------------------------------------------------------------------------
// Mamba layer forward for MI455X (gfx1250, wave32, WMMA bf16 path)
//   D_MODEL=2048, D_INNER=4096, D_STATE=16, D_CONV=4, DT_RANK=128, B=2, T=1024
// ---------------------------------------------------------------------------

#define D_MODEL 2048
#define D_INNER 4096
#define D_STATE 16
#define D_CONV  4
#define DT_RANK 128
#define BB      2
#define TT      1024
#define BT      (BB * TT)           // 2048 rows for all GEMMs

typedef __bf16  bf16_t;
typedef bf16_t  bf16x2  __attribute__((ext_vector_type(2)));
typedef bf16_t  bf16x16 __attribute__((ext_vector_type(16)));
typedef float   f32x8   __attribute__((ext_vector_type(8)));

// ---- helpers --------------------------------------------------------------

__device__ __forceinline__ unsigned cvt_pk_bf16(float lo, float hi) {
#if __has_builtin(__builtin_amdgcn_cvt_pk_bf16_f32)
  // single v_cvt_pk_bf16_f32
  union { bf16x2 v; unsigned u; } r;
  r.v = __builtin_amdgcn_cvt_pk_bf16_f32(lo, hi);
  return r.u;
#else
  // native bf16 truncation; clang lowers to hardware cvt on gfx1250
  union { bf16_t h[2]; unsigned u; } r;
  r.h[0] = (bf16_t)lo;
  r.h[1] = (bf16_t)hi;
  return r.u;
#endif
}

__device__ __forceinline__ float siluf(float x) {
  return x / (1.f + __expf(-x));
}

__device__ __forceinline__ float softplusf(float x) {
  return (x > 20.f) ? x : log1pf(__expf(x));
}

union FragU {
  uint4   u[2];
  bf16x16 v;
};

// load one 16x32 bf16 fragment (A or B operand) from an LDS tile stored as
// 128 rows x 16 dwords (32 bf16) per row.  ISA 16-bit A layout:
//   lane l, m = l&15, hi = l>>4
//   VGPR 0..3 hold K = hi*8 + {0,1 .. 6,7};  VGPR 4..7 hold K = 16 + hi*8 + ...
__device__ __forceinline__ bf16x16 load_frag(const unsigned* lds, int r0, int hi) {
  FragU f;
  f.u[0] = *(const uint4*)&lds[r0 * 16 + hi * 4];
  f.u[1] = *(const uint4*)&lds[r0 * 16 + 8 + hi * 4];
  return f.v;
}

// stage a 128 x 32 fp32 tile (rows rBase.., cols kBase..) as bf16 into LDS
template <bool GUARD>
__device__ __forceinline__ void stage_tile(const float* __restrict__ src, int ld,
                                           int rBase, int RMax, int kBase,
                                           unsigned* lds, int tid) {
#pragma unroll
  for (int i = 0; i < 4; ++i) {
    int idx = tid + i * 256;          // 0..1023 float4 slots
    int r   = idx >> 3;               // 0..127
    int kq  = idx & 7;                // float4 index within the 32-wide row
    int gr  = rBase + r;
    float4 v;
    if (GUARD) {
      v = make_float4(0.f, 0.f, 0.f, 0.f);
      if (gr < RMax)
        v = *(const float4*)(src + (size_t)gr * ld + kBase + kq * 4);
    } else {
      v = *(const float4*)(src + (size_t)gr * ld + kBase + kq * 4);
    }
    uint2 p;
    p.x = cvt_pk_bf16(v.x, v.y);
    p.y = cvt_pk_bf16(v.z, v.w);
    *(uint2*)&lds[r * 16 + kq * 2] = p;
  }
}

// ---- generic bf16 WMMA GEMM: C[m,n] = epi( sum_k A[m,k] * W[n,k] ) --------
// block tile 128x128, 256 threads = 8 waves, wave tile 32(M) x 64(N)
// EPI: 0 = none, 1 = softplus(acc + bias[n])
// GUARD: bounds-check N (only needed when N % 128 != 0)
template <int EPI, bool GUARD>
__global__ __launch_bounds__(256) void gemm_bf16_wmma(
    const float* __restrict__ A, int lda,
    const float* __restrict__ W, int ldw,
    float* __restrict__ C, int ldc,
    int M, int N, int K,
    const float* __restrict__ bias) {
  const int tid   = threadIdx.x;
  const int lane  = tid & 31;          // wave32
  const int wave  = tid >> 5;          // 0..7
  const int waveM = wave & 3;          // 4 waves along M -> *32
  const int waveN = wave >> 2;         // 2 waves along N -> *64
  const int mBase = blockIdx.y * 128;
  const int nBase = blockIdx.x * 128;
  const int l15   = lane & 15;
  const int hi    = lane >> 4;

  __shared__ __align__(16) unsigned lds_a[128 * 16];
  __shared__ __align__(16) unsigned lds_b[128 * 16];

  f32x8 acc[2][4] = {};

  for (int kB = 0; kB < K; kB += 32) {
    stage_tile<false>(A, lda, mBase, M, kB, lds_a, tid);
    stage_tile<GUARD>(W, ldw, nBase, N, kB, lds_b, tid);
    if (kB + 32 < K) {  // speculative prefetch of the next K tiles
      __builtin_prefetch(A + (size_t)(mBase + (tid >> 1)) * lda + kB + 32, 0, 1);
      __builtin_prefetch(W + (size_t)(nBase + (tid >> 1)) * ldw + kB + 32, 0, 1);
    }
    __syncthreads();

    bf16x16 af[2], bf[4];
#pragma unroll
    for (int mt = 0; mt < 2; ++mt)
      af[mt] = load_frag(lds_a, waveM * 32 + mt * 16 + l15, hi);
#pragma unroll
    for (int nt = 0; nt < 4; ++nt)
      bf[nt] = load_frag(lds_b, waveN * 64 + nt * 16 + l15, hi);

#pragma unroll
    for (int mt = 0; mt < 2; ++mt)
#pragma unroll
      for (int nt = 0; nt < 4; ++nt)
        acc[mt][nt] = __builtin_amdgcn_wmma_f32_16x16x32_bf16(
            false, af[mt], false, bf[nt], (short)0, acc[mt][nt], false, false);

    __syncthreads();
  }

  // store: C/D layout -> lane l: n = l&15, m = vgpr + 8*(l>>4)
#pragma unroll
  for (int mt = 0; mt < 2; ++mt) {
#pragma unroll
    for (int nt = 0; nt < 4; ++nt) {
      int gn = nBase + waveN * 64 + nt * 16 + l15;
      if (!GUARD || gn < N) {
#pragma unroll
        for (int v = 0; v < 8; ++v) {
          int gm = mBase + waveM * 32 + mt * 16 + v + 8 * hi;
          float r = acc[mt][nt][v];
          if (EPI == 1) r = softplusf(r + bias[gn]);
          C[(size_t)gm * ldc + gn] = r;
        }
      }
    }
  }
}

// ---- depthwise causal conv(4) + SiLU, also emits new conv cache -----------
__global__ __launch_bounds__(256) void conv_silu_kernel(
    const float* __restrict__ xz,        // (B,T,2*D_INNER), xi = [:, :, :4096]
    const float* __restrict__ cache,     // (B,D_INNER,3)
    const float* __restrict__ cw,        // (D_INNER,4)
    const float* __restrict__ cb,        // (D_INNER)
    float* __restrict__ xs,              // (B,T,D_INNER)
    float* __restrict__ new_cache) {     // (B,D_INNER,3)
  int idx = blockIdx.x * blockDim.x + threadIdx.x;
  if (idx >= BB * TT * D_INNER) return;
  int d = idx & (D_INNER - 1);
  int t = (idx >> 12) & (TT - 1);
  int b = idx >> 22;

  float acc = cb[d];
#pragma unroll
  for (int k = 0; k < D_CONV; ++k) {
    int tt = t - (D_CONV - 1) + k;
    float v;
    if (tt < 0)
      v = cache[((size_t)b * D_INNER + d) * (D_CONV - 1) + (tt + D_CONV - 1)];
    else
      v = xz[((size_t)(b * TT + tt)) * (2 * D_INNER) + d];
    acc += v * cw[d * D_CONV + k];
  }
  xs[((size_t)(b * TT + t)) * D_INNER + d] = siluf(acc);

  if (t >= TT - (D_CONV - 1)) {
    new_cache[((size_t)b * D_INNER + d) * (D_CONV - 1) + (t - (TT - (D_CONV - 1)))] =
        xz[((size_t)(b * TT + t)) * (2 * D_INNER) + d];
  }
}

// ---- selective scan: one thread per (b,d), h[16] in registers -------------
__global__ __launch_bounds__(256) void scan_kernel(
    const float* __restrict__ delta,     // (B,T,D_INNER)
    const float* __restrict__ dbc,       // (B,T,160): [.,128:144]=B, [.,144:160]=C
    const float* __restrict__ xs,        // (B,T,D_INNER)
    const float* __restrict__ A_log,     // (D_INNER,16)
    const float* __restrict__ D_param,   // (D_INNER)
    const float* __restrict__ h_init,    // (B,D_INNER,16)
    float* __restrict__ y,               // (B,T,D_INNER)
    float* __restrict__ h_out) {         // (B,D_INNER,16)
  const int tid = threadIdx.x;
  const int b   = blockIdx.x >> 4;                 // 16 blocks per batch
  const int d   = ((blockIdx.x & 15) << 8) + tid;  // 0..4095

  __shared__ float sBC[32];                        // B_t (16) then C_t (16)

  float An[D_STATE], h[D_STATE];
#pragma unroll
  for (int n = 0; n < D_STATE; ++n) {
    An[n] = -__expf(A_log[d * D_STATE + n]);
    h[n]  = h_init[((size_t)b * D_INNER + d) * D_STATE + n];
  }
  const float Dp = D_param[d];

  for (int t = 0; t < TT; ++t) {
    const size_t bt = (size_t)(b * TT + t);
    __syncthreads();
    if (tid < 32) sBC[tid] = dbc[bt * (DT_RANK + 2 * D_STATE) + DT_RANK + tid];
    __syncthreads();

    float dl = delta[bt * D_INNER + d];
    float xv = xs[bt * D_INNER + d];
    float dx = dl * xv;
    float yacc = 0.f;
#pragma unroll
    for (int n = 0; n < D_STATE; ++n) {
      float dA = __expf(dl * An[n]);
      h[n] = dA * h[n] + dx * sBC[n];
      yacc += h[n] * sBC[D_STATE + n];
    }
    y[bt * D_INNER + d] = yacc + Dp * xv;
  }

#pragma unroll
  for (int n = 0; n < D_STATE; ++n)
    h_out[((size_t)b * D_INNER + d) * D_STATE + n] = h[n];
}

// ---- gate: y *= silu(z) (in place) ----------------------------------------
__global__ __launch_bounds__(256) void gate_kernel(
    float* __restrict__ y, const float* __restrict__ xz) {
  int idx = blockIdx.x * blockDim.x + threadIdx.x;
  if (idx >= BB * TT * D_INNER) return;
  int d = idx & (D_INNER - 1);
  size_t bt = (size_t)(idx >> 12);
  float z = xz[bt * (2 * D_INNER) + D_INNER + d];
  y[idx] = y[idx] * siluf(z);
}

// ---------------------------------------------------------------------------

extern "C" void kernel_launch(void* const* d_in, const int* in_sizes, int n_in,
                              void* d_out, int out_size, void* d_ws, size_t ws_size,
                              hipStream_t stream) {
  const float* x         = (const float*)d_in[0];   // (B,T,D_MODEL)
  const float* h0        = (const float*)d_in[1];   // (B,D_INNER,16)
  const float* conv_c    = (const float*)d_in[2];   // (B,D_INNER,3)
  const float* in_proj_w = (const float*)d_in[3];   // (8192,2048)
  const float* conv_w    = (const float*)d_in[4];   // (4096,4)
  const float* conv_b    = (const float*)d_in[5];   // (4096)
  const float* x_proj_w  = (const float*)d_in[6];   // (160,4096)
  const float* dt_proj_w = (const float*)d_in[7];   // (4096,128)
  const float* dt_proj_b = (const float*)d_in[8];   // (4096)
  const float* A_log     = (const float*)d_in[9];   // (4096,16)
  const float* D_param   = (const float*)d_in[10];  // (4096)
  const float* out_proj_w= (const float*)d_in[11];  // (2048,4096)

  float* out   = (float*)d_out;                                  // (B,T,2048)
  float* h_out = out + (size_t)BB * TT * D_MODEL;                // (B,4096,16)
  float* cache_out = h_out + (size_t)BB * D_INNER * D_STATE;     // (B,4096,3)

  // workspace layout (floats)
  float* ws   = (float*)d_ws;
  float* xz    = ws;                                             // 16,777,216
  float* xs    = xz    + (size_t)BB * TT * 2 * D_INNER;          //  8,388,608
  float* dbc   = xs    + (size_t)BB * TT * D_INNER;              //    327,680
  float* delta = dbc   + (size_t)BB * TT * (DT_RANK + 2 * D_STATE);
  float* y     = delta + (size_t)BB * TT * D_INNER;              //  8,388,608

  dim3 blk(256);

  // 1) xz = x @ in_proj_w.T   (M=2048, K=2048, N=8192)
  gemm_bf16_wmma<0, false><<<dim3(8192 / 128, BT / 128), blk, 0, stream>>>(
      x, D_MODEL, in_proj_w, D_MODEL, xz, 2 * D_INNER,
      BT, 2 * D_INNER, D_MODEL, nullptr);

  // 2) depthwise conv + SiLU (also writes new conv cache)
  conv_silu_kernel<<<(BB * TT * D_INNER + 255) / 256, blk, 0, stream>>>(
      xz, conv_c, conv_w, conv_b, xs, cache_out);

  // 3) dbc = xs @ x_proj_w.T  (M=2048, K=4096, N=160)  -- needs N guard
  gemm_bf16_wmma<0, true><<<dim3(2, BT / 128), blk, 0, stream>>>(
      xs, D_INNER, x_proj_w, D_INNER, dbc, DT_RANK + 2 * D_STATE,
      BT, DT_RANK + 2 * D_STATE, D_INNER, nullptr);

  // 4) delta = softplus(dbc[:, :128] @ dt_proj_w.T + b)  (M=2048, K=128, N=4096)
  gemm_bf16_wmma<1, false><<<dim3(4096 / 128, BT / 128), blk, 0, stream>>>(
      dbc, DT_RANK + 2 * D_STATE, dt_proj_w, DT_RANK, delta, D_INNER,
      BT, D_INNER, DT_RANK, dt_proj_b);

  // 5) selective scan (writes y and h_last)
  scan_kernel<<<BB * (D_INNER / 256), blk, 0, stream>>>(
      delta, dbc, xs, A_log, D_param, h0, y, h_out);

  // 6) y *= silu(z)
  gate_kernel<<<(BB * TT * D_INNER + 255) / 256, blk, 0, stream>>>(y, xz);

  // 7) out = y @ out_proj_w.T  (M=2048, K=4096, N=2048)
  gemm_bf16_wmma<0, false><<<dim3(2048 / 128, BT / 128), blk, 0, stream>>>(
      y, D_INNER, out_proj_w, D_INNER, out, D_MODEL,
      BT, D_MODEL, D_INNER, nullptr);
}
